// MultiHeadAttention_51470888075437
// MI455X (gfx1250) — compile-verified, matrix operand order checked by reading, not run on hardware
//
#include <hip/hip_runtime.h>
#include <hip/hip_bf16.h>

// MI455X / gfx1250, wave32. bf16 WMMA (v_wmma_f32_16x16x32_bf16) + async
// global->LDS staging (GLOBAL_LOAD_ASYNC_TO_LDS_B128, ASYNCcnt).
typedef __bf16 bf16;
typedef __attribute__((ext_vector_type(16))) __bf16 v16bf;
typedef __attribute__((ext_vector_type(4)))  __bf16 v4bf;
typedef __attribute__((ext_vector_type(8)))  float  v8f;
typedef __attribute__((ext_vector_type(4)))  int    v4i;

typedef __attribute__((address_space(1))) v4i* gv4i_t;  // global int4*
typedef __attribute__((address_space(3))) v4i* lv4i_t;  // LDS int4*
typedef __attribute__((address_space(3))) char* lds_ptr_t;
typedef __attribute__((address_space(1))) char* gbl_ptr_t;

#define WMMA_BF16(a, b, c) \
  __builtin_amdgcn_wmma_f32_16x16x32_bf16(false, (a), false, (b), (short)0, (c), false, false)

// ---- Async global->LDS 16B copy (per-lane), tracked by ASYNCcnt -------------
__device__ __forceinline__ void async_b128(const void* g, void* l) {
#if __has_builtin(__builtin_amdgcn_global_load_async_to_lds_b128)
  __builtin_amdgcn_global_load_async_to_lds_b128((gv4i_t)g, (lv4i_t)l, 0, 0);
#else
  asm volatile("global_load_async_to_lds_b128 %0, %1, off"
               :: "v"((lds_ptr_t)l), "v"((gbl_ptr_t)g)
               : "memory");
#endif
}

template <int N>
__device__ __forceinline__ void wait_async() {
#if __has_builtin(__builtin_amdgcn_s_wait_asynccnt)
  __builtin_amdgcn_s_wait_asynccnt(N);
#else
  asm volatile("s_wait_asynccnt %0" :: "n"(N) : "memory");
#endif
}

// ---- WMMA fragment helpers (CDNA5 ISA 7.12.2 layouts) -----------------------
// 16-bit A 16x32: lanes 0-15 -> M=lane, K in {0..7,16..23}; lanes 16-31 ->
// M=lane-16, K in {8..15,24..31}; each VGPR packs 2 consecutive K.
__device__ __forceinline__ int kmap(int v, int half) {
  int base = (v < 4) ? (2 * v) : (16 + 2 * (v - 4));
  return base + (half ? 8 : 0);
}

// A fragment (16x32) from bf16 LDS tile, row-major. Also the B fragment for
// S = Q*K^T since B[k][n] = Ktile[n][k] has the identical access pattern.
__device__ __forceinline__ v16bf frag_row_bf16(const bf16* t, int ld, int row0, int k0, int lane) {
  int half = lane >> 4, r = row0 + (lane & 15);
  v16bf f;
#pragma unroll
  for (int v = 0; v < 8; ++v) {
    int k = k0 + kmap(v, half);
    f[2 * v]     = t[r * ld + k];
    f[2 * v + 1] = t[r * ld + k + 1];
  }
  return f;
}

// A fragment from f32 LDS tile (converted to bf16 in-register).
__device__ __forceinline__ v16bf fragA_row_f32(const float* t, int ld, int row0, int k0, int lane) {
  int half = lane >> 4, r = row0 + (lane & 15);
  v16bf f;
#pragma unroll
  for (int v = 0; v < 8; ++v) {
    int k = k0 + kmap(v, half);
    f[2 * v]     = (bf16)t[r * ld + k];
    f[2 * v + 1] = (bf16)t[r * ld + k + 1];
  }
  return f;
}

// B fragment (32x16) where B[k][n] = tile[k][col0+n] (tile row-major over k).
__device__ __forceinline__ v16bf fragB_col_bf16(const bf16* t, int ld, int k0, int col0, int lane) {
  int half = lane >> 4, c = col0 + (lane & 15);
  v16bf f;
#pragma unroll
  for (int v = 0; v < 8; ++v) {
    int k = k0 + kmap(v, half);
    f[2 * v]     = t[k * ld + c];
    f[2 * v + 1] = t[(k + 1) * ld + c];
  }
  return f;
}

// ---- Generic tiled GEMM + bias: Out[RxN] = X[RxK] * W[KxN] + bias -----------
// 512 threads = 16 waves, each wave owns one 16x16 tile of a 64x64 block.
// Staging: one 16B global load + one 8B ds store per thread per tile.
template <bool XBF16, bool OUTBF16>
__global__ __launch_bounds__(512) void gemm_bias(const void* __restrict__ Xv,
                                                 const float* __restrict__ W,
                                                 const float* __restrict__ bias,
                                                 void* __restrict__ Outv,
                                                 int Kdim, int ldx, int ldw, int ldo) {
  __shared__ bf16 Xt[64 * 36];  // 64 rows x 32 k (stride 36: 8B-aligned rows)
  __shared__ bf16 Wt[32 * 72];  // 32 k x 64 cols (stride 72)
  const int tid = threadIdx.x, wave = tid >> 5, lane = tid & 31;
  const int r0 = blockIdx.y * 64, c0 = blockIdx.x * 64;
  const int tr = (wave >> 2) * 16, tc = (wave & 3) * 16;
  const float* Xf = (const float*)Xv;
  const bf16*  Xb = (const bf16*)Xv;
  // staging coordinates
  const int xr = tid >> 3, xq = (tid & 7) * 4;    // X: row 0..63, 4-elem group
  const int wr = tid >> 4, wq = (tid & 15) * 4;   // W: k 0..31, 4-elem group
  v8f acc = {0, 0, 0, 0, 0, 0, 0, 0};
  for (int kt = 0; kt < Kdim; kt += 32) {
    v4bf xe;
    if (XBF16) {
      xe = *(const v4bf*)(Xb + (size_t)(r0 + xr) * ldx + kt + xq);
    } else {
      float4 v = *(const float4*)(Xf + (size_t)(r0 + xr) * ldx + kt + xq);
      xe = (v4bf){(bf16)v.x, (bf16)v.y, (bf16)v.z, (bf16)v.w};
    }
    float4 wv = *(const float4*)(W + (size_t)(kt + wr) * ldw + c0 + wq);
    *(v4bf*)(Xt + xr * 36 + xq) = xe;
    *(v4bf*)(Wt + wr * 72 + wq) = (v4bf){(bf16)wv.x, (bf16)wv.y, (bf16)wv.z, (bf16)wv.w};
    __syncthreads();
    v16bf a = frag_row_bf16(Xt, 36, tr, 0, lane);
    v16bf b = fragB_col_bf16(Wt, 72, 0, tc, lane);
    acc = WMMA_BF16(a, b, acc);
    __syncthreads();
  }
  const int half = lane >> 4, c = tc + (lane & 15);
  const float bv = bias[c0 + c];
#pragma unroll
  for (int i = 0; i < 8; ++i) {
    size_t r = (size_t)(r0 + tr + i + 8 * half);
    float vout = acc[i] + bv;
    if (OUTBF16) ((bf16*)Outv)[r * ldo + c0 + c] = (bf16)vout;
    else         ((float*)Outv)[r * ldo + c0 + c] = vout;
  }
}

// ---- Fused attention: per (b, h, 64-row q tile) -----------------------------
// 64x1024 f32 score strip lives in LDS (256KB of 320KB/WGP); K/V tiles are
// double-buffered via async global->LDS copies overlapped with WMMA compute.
__global__ __launch_bounds__(256) void attention_fused(const bf16* __restrict__ Qg,
                                                       const bf16* __restrict__ KVg,
                                                       const unsigned char* __restrict__ mask,
                                                       float* __restrict__ att,
                                                       bf16* __restrict__ attV) {
  constexpr int Nc = 1024, Mc = 1024, Cc = 1024, Hc = 16, Dc = 64, TT = 64;
  constexpr float SCALE = 0.125f;  // D^-0.5, D=64
  extern __shared__ char smem[];
  float* S   = (float*)smem;                          // 64 x 1024 f32 (256KB)
  bf16* Qt   = (bf16*)(smem + (size_t)TT * Mc * 4);   // 64 x 72 bf16
  bf16* KV0  = Qt  + TT * 72;                         // 64 x 72 bf16 (buf 0)
  bf16* KV1  = KV0 + TT * 72;                         // 64 x 72 bf16 (buf 1)
  const int tid = threadIdx.x, wave = tid >> 5, lane = tid & 31;
  const int bid = blockIdx.x;
  const int nt = bid & 15, h = (bid >> 4) & 15, b = bid >> 8;
  const int n0 = nt * TT;
  const int trow = (wave & 3) * 16;   // wave's output row tile
  const int tcb  = (wave >> 2) * 16;  // wave's first column tile (pairs with +32)
  // async staging coords: 2 x (row, 16B chunk) per thread for a 64x64 bf16 tile
  const int sr0 = tid >> 3,         sc0 = (tid & 7) * 8;
  const int sr1 = (tid + 256) >> 3, sc1 = ((tid + 256) & 7) * 8;

  const bf16* Qsrc = Qg + ((size_t)b * Nc + n0) * Cc + (size_t)h * Dc;
  const bf16* Ksrc = KVg + (size_t)b * Mc * (2 * Cc) + (size_t)h * Dc;  // stride 2C
  const bf16* Vsrc = Ksrc + Cc;                                         // stride 2C

  // Prologue: async-stage Q tile and K chunks 0,1.
  async_b128(Qsrc + (size_t)sr0 * Cc + sc0, Qt + sr0 * 72 + sc0);
  async_b128(Qsrc + (size_t)sr1 * Cc + sc1, Qt + sr1 * 72 + sc1);
  async_b128(Ksrc + (size_t)sr0 * (2 * Cc) + sc0, KV0 + sr0 * 72 + sc0);
  async_b128(Ksrc + (size_t)sr1 * (2 * Cc) + sc1, KV0 + sr1 * 72 + sc1);
  async_b128(Ksrc + (size_t)(TT + sr0) * (2 * Cc) + sc0, KV1 + sr0 * 72 + sc0);
  async_b128(Ksrc + (size_t)(TT + sr1) * (2 * Cc) + sc1, KV1 + sr1 * 72 + sc1);

  // Phase 1: S = SCALE * Q K^T  (64 x 1024 into LDS), K double-buffered.
  for (int mc = 0; mc < 16; ++mc) {
    if (mc < 15) wait_async<2>(); else wait_async<0>();  // in-order: current buf done
    __syncthreads();
    const bf16* Kt = (mc & 1) ? KV1 : KV0;
#pragma unroll
    for (int t = 0; t < 2; ++t) {
      const int tc = tcb + t * 32;
      v8f acc = {0, 0, 0, 0, 0, 0, 0, 0};
#pragma unroll
      for (int k0 = 0; k0 < Dc; k0 += 32) {
        v16bf a  = frag_row_bf16(Qt, 72, trow, k0, lane);
        v16bf bb = frag_row_bf16(Kt, 72, tc, k0, lane);  // B[k][m] = K[m][k]
        acc = WMMA_BF16(a, bb, acc);
      }
      const int half = lane >> 4, c = lane & 15;
#pragma unroll
      for (int i = 0; i < 8; ++i)
        S[(size_t)(trow + i + 8 * half) * Mc + mc * TT + tc + c] = acc[i] * SCALE;
    }
    __syncthreads();  // all waves done with this K buffer
    if (mc + 2 < 16) {
      bf16* nb = (mc & 1) ? KV1 : KV0;
      async_b128(Ksrc + (size_t)((mc + 2) * TT + sr0) * (2 * Cc) + sc0, nb + sr0 * 72 + sc0);
      async_b128(Ksrc + (size_t)((mc + 2) * TT + sr1) * (2 * Cc) + sc1, nb + sr1 * 72 + sc1);
    }
  }

  // Prefetch first two V chunks; their latency hides under the softmax phase.
  async_b128(Vsrc + (size_t)sr0 * (2 * Cc) + sc0, KV0 + sr0 * 72 + sc0);
  async_b128(Vsrc + (size_t)sr1 * (2 * Cc) + sc1, KV0 + sr1 * 72 + sc1);
  async_b128(Vsrc + (size_t)(TT + sr0) * (2 * Cc) + sc0, KV1 + sr0 * 72 + sc0);
  async_b128(Vsrc + (size_t)(TT + sr1) * (2 * Cc) + sc1, KV1 + sr1 * 72 + sc1);

  // Phase 2: masked softmax per row; write normalized att to global.
  {
    const unsigned char* mrow = mask + (size_t)b * Mc;
    for (int j = 0; j < 8; ++j) {
      const int r = wave * 8 + j;
      float* Sr = S + (size_t)r * Mc;
      float mx = -__builtin_inff();
      for (int m = lane; m < Mc; m += 32) {
        float v = mrow[m] ? Sr[m] : -__builtin_inff();
        Sr[m] = v;
        mx = fmaxf(mx, v);
      }
      for (int o = 16; o > 0; o >>= 1) mx = fmaxf(mx, __shfl_xor(mx, o, 32));
      float sum = 0.f;
      for (int m = lane; m < Mc; m += 32) {
        float e = __expf(Sr[m] - mx);
        Sr[m] = e;
        sum += e;
      }
      for (int o = 16; o > 0; o >>= 1) sum += __shfl_xor(sum, o, 32);
      const float inv = 1.0f / sum;
      float* arow = att + (((size_t)b * Nc + n0 + r) * Mc) * Hc + h;  // att[b,n,m,h]
      for (int m = lane; m < Mc; m += 32) {
        float p = Sr[m] * inv;
        Sr[m] = p;
        arow[(size_t)m * Hc] = p;
      }
    }
  }

  // Phase 3: O = P @ V (fp32 accumulate over 16 m-chunks), V double-buffered.
  v8f accO0 = {0, 0, 0, 0, 0, 0, 0, 0};
  v8f accO1 = {0, 0, 0, 0, 0, 0, 0, 0};
  for (int mc = 0; mc < 16; ++mc) {
    if (mc < 15) wait_async<2>(); else wait_async<0>();
    __syncthreads();
    const bf16* Vt = (mc & 1) ? KV1 : KV0;
#pragma unroll
    for (int k0 = 0; k0 < TT; k0 += 32) {
      v16bf a  = fragA_row_f32(S, Mc, trow, mc * TT + k0, lane);
      v16bf b0 = fragB_col_bf16(Vt, 72, k0, tcb, lane);
      accO0 = WMMA_BF16(a, b0, accO0);
      v16bf b1 = fragB_col_bf16(Vt, 72, k0, tcb + 32, lane);
      accO1 = WMMA_BF16(a, b1, accO1);
    }
    __syncthreads();  // all waves done with this V buffer
    if (mc + 2 < 16) {
      bf16* nb = (mc & 1) ? KV1 : KV0;
      async_b128(Vsrc + (size_t)((mc + 2) * TT + sr0) * (2 * Cc) + sc0, nb + sr0 * 72 + sc0);
      async_b128(Vsrc + (size_t)((mc + 2) * TT + sr1) * (2 * Cc) + sc1, nb + sr1 * 72 + sc1);
    }
  }
  {
    const int half = lane >> 4, c = lane & 15;
#pragma unroll
    for (int i = 0; i < 8; ++i) {
      size_t row = (size_t)b * Nc + n0 + trow + i + 8 * half;
      attV[row * Cc + h * Dc + tcb + c]      = (bf16)accO0[i];
      attV[row * Cc + h * Dc + tcb + 32 + c] = (bf16)accO1[i];
    }
  }
}

extern "C" void kernel_launch(void* const* d_in, const int* in_sizes, int n_in,
                              void* d_out, int out_size, void* d_ws, size_t ws_size,
                              hipStream_t stream) {
  (void)in_sizes; (void)n_in; (void)out_size; (void)ws_size;
  // B=4, N=M=C=1024, H=16, D=64
  const float* x    = (const float*)d_in[0];
  const float* y    = (const float*)d_in[1];
  const unsigned char* mask = (const unsigned char*)d_in[2];  // jnp bool -> 1 byte
  const float* Wq   = (const float*)d_in[3];
  const float* bq   = (const float*)d_in[4];
  const float* Wkv  = (const float*)d_in[5];
  const float* bkv  = (const float*)d_in[6];
  const float* Wp   = (const float*)d_in[7];
  const float* bp   = (const float*)d_in[8];

  float* out = (float*)d_out;                  // (B,N,C) f32
  float* att = out + (size_t)4 * 1024 * 1024;  // (B,N,M,H) f32

  // Workspace: Qbf 8MB | KVbf 16MB | attVbf 8MB (all bf16)
  bf16* Qbf   = (bf16*)d_ws;
  bf16* KVbf  = Qbf + (size_t)4096 * 1024;
  bf16* attVb = KVbf + (size_t)4096 * 2048;

  // 1) Q = x*Wq + bq (4096x1024, bf16 out)
  gemm_bias<false, true><<<dim3(16, 64), 512, 0, stream>>>(
      x, Wq, bq, Qbf, 1024, 1024, 1024, 1024);
  // 2) KV = y*Wkv + bkv (4096x2048, bf16 out); K = cols [0,1024), V = [1024,2048)
  gemm_bias<false, true><<<dim3(32, 64), 512, 0, stream>>>(
      y, Wkv, bkv, KVbf, 1024, 1024, 2048, 2048);

  // 3) Fused attention: B*H*(N/64) = 1024 blocks, 256 threads, ~284KB dynamic LDS.
  const size_t shmem = (size_t)64 * 1024 * 4 + 3 * (size_t)64 * 72 * 2;
  (void)hipFuncSetAttribute(reinterpret_cast<const void*>(attention_fused),
                            hipFuncAttributeMaxDynamicSharedMemorySize, (int)shmem);
  attention_fused<<<dim3(1024), 256, shmem, stream>>>(Qbf, KVbf, mask, att, attVb);

  // 4) out = attV*Wp + bp (f32 out)
  gemm_bias<true, false><<<dim3(16, 64), 512, 0, stream>>>(
      attVb, Wp, bp, out, 1024, 1024, 1024, 1024);
}
